// ScaledDotProductAttention_26731876450934
// MI455X (gfx1250) — matrix operand order repaired and checked by reading, not compiled, not run on hardware
//
#include <hip/hip_runtime.h>
#include <stdint.h>

typedef __attribute__((ext_vector_type(16))) _Float16 v16h;
typedef __attribute__((ext_vector_type(8)))  float    v8f;
typedef __attribute__((ext_vector_type(4)))  float    v4f;   // native vector (nontemporal-builtin compatible)

#define B_   2
#define H_   16
#define S_   2048
#define D_   64
#define QT   16            // query rows per block (one WMMA M-tile, shared by 4 waves)
#define SSTR (S_ + 8)      // padded LDS row stride in floats (16B-aligned, staggers banks)

// A-matrix (16x32 f16) K-index for packed element e, lane-group g (contiguous: k = e + 16*g, matches B)
__device__ __forceinline__ int a_kidx(int e, int g) {
    return e + 16 * g;
}

__global__ __launch_bounds__(128) void ScaledDotProductAttention_26731876450934_kernel(
        const float* __restrict__ Q, const float* __restrict__ K,
        const float* __restrict__ V, const uint8_t* __restrict__ Msk,
        float* __restrict__ Out, float* __restrict__ W)
{
    __shared__ float sm[QT * SSTR];     // 16 x 2048 score strip (padded), ~128.5 KB of 320 KB WGP LDS
    __shared__ float red[QT * 8];
    __shared__ float rowmax[QT];
    __shared__ float rowinv[QT];

    const int tid  = threadIdx.x;
    const int lane = tid & 31;
    const int wv   = tid >> 5;          // wave id 0..3
    const int lm   = lane & 15;         // lane % 16
    const int lg   = lane >> 4;         // lane group 0/1

    const int bid = blockIdx.x;
    const int qt  = bid & 127;          // S_/QT = 128 query tiles
    const int h   = (bid >> 7) & 15;
    const int b   = bid >> 11;
    const int q0  = qt * QT;
    const size_t bh = (size_t)(b * H_ + h);

    const float scale = 0.125f;         // 1/sqrt(64)

    // ---------------- Pass A: scores = scale * Q K^T, masked -> LDS ----------------
    // Q tile 16x64 as two 16x32 f16 A-fragments (per-lane contiguous 16-float chunks)
    const float* qrow = Q + (bh * S_ + (size_t)(q0 + lm)) * D_;
    v16h qa0, qa1;
#pragma unroll
    for (int e = 0; e < 16; ++e) {
        qa0[e] = (_Float16)qrow[a_kidx(e, lg)];
        qa1[e] = (_Float16)qrow[32 + a_kidx(e, lg)];
    }

    for (int t = 0; t < 32; ++t) {
        const int colbase = wv * 512 + t * 16;
        // B-fragment of K^T (D x 16): N = lm -> K row index; K-dim d = e + 16*lg
        const float* krp = K + (bh * S_ + (size_t)(colbase + lm)) * D_;
        v16h kb0, kb1;
#pragma unroll
        for (int e = 0; e < 16; ++e) {
            kb0[e] = (_Float16)krp[e + 16 * lg];
            kb1[e] = (_Float16)krp[32 + e + 16 * lg];
        }
        v8f c = {};
        c = __builtin_amdgcn_wmma_f32_16x16x32_f16(false, qa0, false, kb0, (short)0, c, false, false);
        c = __builtin_amdgcn_wmma_f32_16x16x32_f16(false, qa1, false, kb1, (short)0, c, false, false);

        const int col = colbase + lm;   // C/D: N = lm, M = v + 8*lg
#pragma unroll
        for (int v = 0; v < 8; ++v) {
            const int r = v + 8 * lg;
            float s = c[v] * scale;
            if (Msk[((size_t)b * S_ + (size_t)(q0 + r)) * S_ + col]) s = -__builtin_inff();
            sm[r * SSTR + col] = s;
        }
    }
    __syncthreads();

    // ---------------- Softmax: row max (float4-vectorized LDS reads) ----------------
    {
        const int row = tid >> 3, sub = tid & 7;
        const v4f* p = (const v4f*)&sm[row * SSTR + sub * 256];
        float m = -__builtin_inff();
        for (int i = 0; i < 64; ++i) {
            v4f v4 = p[i];
            m = fmaxf(m, fmaxf(fmaxf(v4.x, v4.y), fmaxf(v4.z, v4.w)));
        }
        red[row * 8 + sub] = m;
    }
    __syncthreads();
    if (tid < QT) {
        float m = red[tid * 8];
#pragma unroll
        for (int i = 1; i < 8; ++i) m = fmaxf(m, red[tid * 8 + i]);
        rowmax[tid] = (m == -__builtin_inff()) ? 0.0f : m;
    }
    __syncthreads();

    // ---------------- exp(s - max) in place + row sum (float4-vectorized) ----------------
    {
        const int row = tid >> 3, sub = tid & 7;
        v4f* p = (v4f*)&sm[row * SSTR + sub * 256];
        const float mx = rowmax[row];
        float sum = 0.0f;
        for (int i = 0; i < 64; ++i) {
            v4f v4 = p[i];
            v4.x = __expf(v4.x - mx);
            v4.y = __expf(v4.y - mx);
            v4.z = __expf(v4.z - mx);
            v4.w = __expf(v4.w - mx);
            p[i] = v4;
            sum += (v4.x + v4.y) + (v4.z + v4.w);
        }
        red[row * 8 + sub] = sum;
    }
    __syncthreads();
    if (tid < QT) {
        float s = 0.0f;
#pragma unroll
        for (int i = 0; i < 8; ++i) s += red[tid * 8 + i];
        rowinv[tid] = (s > 0.0f) ? 1.0f / s : 0.0f;
    }
    __syncthreads();

    // --- Normalize P in LDS + coalesced NON-TEMPORAL float4 write of weights ---
    // Weights (512 MB total) are write-once / never re-read: TH=NT keeps K/V/mask
    // (which 128 q-tile WGs per (b,h) re-read) resident in the 192 MB L2.
    {
        float* wbase = W + (bh * S_ + (size_t)q0) * (size_t)S_;
        for (int idx = tid; idx < QT * (S_ / 4); idx += 128) {
            const int row = idx >> 9;              // 512 float4 per row
            const int c4  = idx & 511;
            v4f vv = *(v4f*)&sm[row * SSTR + c4 * 4];
            const float inv = rowinv[row];
            vv.x *= inv; vv.y *= inv; vv.z *= inv; vv.w *= inv;
            *(v4f*)&sm[row * SSTR + c4 * 4] = vv;
            __builtin_nontemporal_store(vv, (v4f*)&wbase[(size_t)row * S_ + c4 * 4]);
        }
    }
    __syncthreads();

    // ---------------- Pass C: Out = P @ V, each wave owns 16 output cols ----------------
    {
        const int n0 = wv * 16;
        v8f acc = {};
        for (int kt = 0; kt < 64; ++kt) {
            const int kb = kt * 32;
            if (kt + 1 < 64) {
                // global_prefetch_b8 of next V block for this lane's column
                __builtin_prefetch(&V[(bh * S_ + (size_t)((kt + 1) * 32 + 16 * lg)) * D_ + n0 + lm], 0, 1);
            }
            v16h pa, vb;
#pragma unroll
            for (int e = 0; e < 16; ++e) {
                pa[e] = (_Float16)sm[lm * SSTR + kb + a_kidx(e, lg)];
                vb[e] = (_Float16)V[(bh * S_ + (size_t)(kb + e + 16 * lg)) * D_ + n0 + lm];
            }
            acc = __builtin_amdgcn_wmma_f32_16x16x32_f16(false, pa, false, vb, (short)0, acc, false, false);
        }
#pragma unroll
        for (int v = 0; v < 8; ++v) {
            const int r = v + 8 * lg;
            __builtin_nontemporal_store(acc[v], &Out[(bh * S_ + (size_t)(q0 + r)) * D_ + n0 + lm]);
        }
    }
}

extern "C" void kernel_launch(void* const* d_in, const int* in_sizes, int n_in,
                              void* d_out, int out_size, void* d_ws, size_t ws_size,
                              hipStream_t stream) {
    (void)in_sizes; (void)n_in; (void)d_ws; (void)ws_size; (void)out_size;
    const float*   Q   = (const float*)d_in[0];
    const float*   K   = (const float*)d_in[1];
    const float*   V   = (const float*)d_in[2];
    const uint8_t* Msk = (const uint8_t*)d_in[3];   // jax bool mask, 1 byte/elem
    float* Out = (float*)d_out;                                  // (B,H,S,D)
    float* W   = Out + (size_t)B_ * H_ * S_ * D_;                // (B,H,S,S)
    dim3 grid(B_ * H_ * (S_ / QT));   // 4096 blocks
    ScaledDotProductAttention_26731876450934_kernel<<<grid, 128, 0, stream>>>(Q, K, V, Msk, Out, W);
}
